// GATLayer_37203006718640
// MI455X (gfx1250) — matrix operand order repaired and checked by reading, not compiled, not run on hardware
//
#include <hip/hip_runtime.h>

#define B_       4
#define N_       4096
#define FIN_     128
#define FOUT_    64
#define ALPHA_   0.2f
#define OUTSL_   0.01f
#define NEGINF_  (-1.0e12f)

typedef __attribute__((ext_vector_type(16))) __bf16 v16bf;
typedef __attribute__((ext_vector_type(8)))  __bf16 v8bf;
typedef __attribute__((ext_vector_type(8)))  float  v8f;
typedef __attribute__((ext_vector_type(4)))  int    v4i;   // native vector for NT loads

// ---------------------------------------------------------------------------
// Kernel 1: h = inp @ W (per node), bf16 copy of h, src = h.a1, dst = h.a2
// One block per (b,n) node, 64 threads (one per output feature).
// ---------------------------------------------------------------------------
__global__ __launch_bounds__(64)
void gat_prep(const float* __restrict__ inp, const float* __restrict__ W,
              const float* __restrict__ a,
              __bf16* __restrict__ hb,
              float* __restrict__ srcv, float* __restrict__ dstv) {
  __shared__ float row[FIN_];
  __shared__ float red[FOUT_];
  const int bn = blockIdx.x;
  const int o  = threadIdx.x;

  row[o]      = inp[bn * FIN_ + o];
  row[o + 64] = inp[bn * FIN_ + 64 + o];
  __syncthreads();

  float acc = 0.f;
#pragma unroll 8
  for (int f = 0; f < FIN_; ++f)
    acc = fmaf(row[f], W[f * FOUT_ + o], acc);   // coalesced W reads (L2 resident)

  hb[bn * FOUT_ + o] = (__bf16)acc;

  // src reduction
  red[o] = acc * a[o];
  __syncthreads();
  for (int s = 32; s > 0; s >>= 1) {
    if (o < s) red[o] += red[o + s];
    __syncthreads();
  }
  if (o == 0) srcv[bn] = red[0];
  __syncthreads();

  // dst reduction
  red[o] = acc * a[FOUT_ + o];
  __syncthreads();
  for (int s = 32; s > 0; s >>= 1) {
    if (o < s) red[o] += red[o + s];
    __syncthreads();
  }
  if (o == 0) dstv[bn] = red[0];
}

// ---------------------------------------------------------------------------
// Kernel 2: fused masked-softmax attention + P@h (flash-attention style).
// Block = 128 threads (4 wave32), 16 rows per block, each wave owns 16 of the
// 64 output features. Streams adj exactly once (NT hint keeps L2 for h).
// Row softmax state (m,l) lives replicated in the 8 producer lanes of each
// row (shfl_xor reductions); only the rescale factor crosses waves via LDS.
// 64 columns (2 WMMAs) per iteration, 2 barriers per iteration.
// ---------------------------------------------------------------------------
__global__ __launch_bounds__(128)
void gat_attn(const int* __restrict__ adj, const __bf16* __restrict__ hb,
              const float* __restrict__ srcv, const float* __restrict__ dstv,
              float* __restrict__ out) {
  __shared__ __bf16 ldsP[16][64];      // P tile (bf16), row-major, 2 KB
  __shared__ float  sscale[16];        // per-row accumulator rescale
  __shared__ float  sdenom[16];        // per-row softmax denominator (final)

  const int t    = threadIdx.x;
  const int b    = blockIdx.y;
  const int rb   = blockIdx.x * 16;    // row block base
  const int prow = t >> 3;             // producer row (0..15), 8 lanes per row
  const int pcg  = t & 7;              // producer col-group (4 cols each)
  const int lane = t & 31;
  const int wave = t >> 5;
  const int hh   = lane >> 4;          // lane half (A/B K-half select)
  const int nn   = lane & 15;
  const int ob   = wave * 16;          // feature base of this wave

  float m_run = -3.0e38f;              // running row max (replicated in 8 lanes)
  float l_run = 0.f;                   // running row denom (replicated)
  v8f   acc   = {};                    // 16x16 fp32 accumulator tile (this wave)

  const float srow    = srcv[b * N_ + rb + prow];
  const long  adjbase = ((long)b * N_ + (rb + prow)) * N_;
  const int   col     = pcg * 4;

  for (int cb = 0; cb < N_; cb += 64) {
    // ---- producers: masked LeakyReLU scores for a 16x64 tile ----
    // NT loads: adj is streamed once, keep L2 resident for hb.
    const v4i    av0 = __builtin_nontemporal_load(
                         (const v4i*)&adj[adjbase + cb + col]);
    const v4i    av1 = __builtin_nontemporal_load(
                         (const v4i*)&adj[adjbase + cb + 32 + col]);
    const float4 dv0 = *(const float4*)&dstv[b * N_ + cb + col];
    const float4 dv1 = *(const float4*)&dstv[b * N_ + cb + 32 + col];
    __builtin_prefetch(&adj[adjbase + cb + col + 4096], 0, 0);

    float s[8];
    {
      const float d[8] = {dv0.x, dv0.y, dv0.z, dv0.w, dv1.x, dv1.y, dv1.z, dv1.w};
#pragma unroll
      for (int k = 0; k < 8; ++k) {
        const int adk = (k < 4) ? av0[k] : av1[k - 4];
        float e = srow + d[k];
        e = e > 0.f ? e : ALPHA_ * e;
        s[k] = adk > 0 ? e : NEGINF_;
      }
    }

    // ---- row max via intra-wave 8-lane shuffle reduction ----
    float tm = s[0];
#pragma unroll
    for (int k = 1; k < 8; ++k) tm = fmaxf(tm, s[k]);
    tm = fmaxf(tm, __shfl_xor(tm, 1));
    tm = fmaxf(tm, __shfl_xor(tm, 2));
    tm = fmaxf(tm, __shfl_xor(tm, 4));

    const float mn = fmaxf(m_run, tm);
    const float sc = __expf(m_run - mn);
    m_run = mn;

    // ---- exponentials -> bf16 P tile + row sum ----
    float ps = 0.f;
#pragma unroll
    for (int k = 0; k < 4; ++k) {
      const float p0 = __expf(s[k]     - mn);
      const float p1 = __expf(s[4 + k] - mn);
      ldsP[prow][col + k]      = (__bf16)p0;
      ldsP[prow][32 + col + k] = (__bf16)p1;
      ps += p0 + p1;
    }
    ps += __shfl_xor(ps, 1);
    ps += __shfl_xor(ps, 2);
    ps += __shfl_xor(ps, 4);
    l_run = l_run * sc + ps;

    if (pcg == 0) sscale[prow] = sc;
    __syncthreads();

    // ---- consumers: rescale accumulator, 2x (load A/B, WMMA) ----
#pragma unroll
    for (int r = 0; r < 8; ++r) acc[r] *= sscale[r + 8 * hh];  // row = r + 8*(lane/16)

    // A (16x32 bf16): lane holds row M=lane%16; K halves {0..7,16..23}/{8..15,24..31}
    {
      const v8bf alo = *(const v8bf*)&ldsP[nn][hh * 8];
      const v8bf ahi = *(const v8bf*)&ldsP[nn][16 + hh * 8];
      v16bf amat;
#pragma unroll
      for (int i = 0; i < 8; ++i) { amat[i] = alo[i]; amat[8 + i] = ahi[i]; }
      // B (32x16 bf16): lane holds N = lane%16; K halves {0..7,16..23}/{8..15,24..31}
      v16bf bmat;
#pragma unroll
      for (int i = 0; i < 8; ++i) {
        bmat[i]     = hb[((long)b * N_ + cb + hh * 8 + i) * FOUT_ + ob + nn];
        bmat[8 + i] = hb[((long)b * N_ + cb + 16 + hh * 8 + i) * FOUT_ + ob + nn];
      }
      acc = __builtin_amdgcn_wmma_f32_16x16x32_bf16(
          false, amat, false, bmat, (short)0, acc, false, false);
    }
    {
      const v8bf alo = *(const v8bf*)&ldsP[nn][32 + hh * 8];
      const v8bf ahi = *(const v8bf*)&ldsP[nn][48 + hh * 8];
      v16bf amat;
#pragma unroll
      for (int i = 0; i < 8; ++i) { amat[i] = alo[i]; amat[8 + i] = ahi[i]; }
      v16bf bmat;
#pragma unroll
      for (int i = 0; i < 8; ++i) {
        bmat[i]     = hb[((long)b * N_ + cb + 32 + hh * 8 + i) * FOUT_ + ob + nn];
        bmat[8 + i] = hb[((long)b * N_ + cb + 48 + hh * 8 + i) * FOUT_ + ob + nn];
      }
      acc = __builtin_amdgcn_wmma_f32_16x16x32_bf16(
          false, amat, false, bmat, (short)0, acc, false, false);
    }
    __syncthreads();   // protect ldsP against next iteration's producers
  }

  // ---- publish denominators, then epilogue ----
  if (pcg == 0) sdenom[prow] = l_run;
  __syncthreads();

#pragma unroll
  for (int r = 0; r < 8; ++r) {
    const int row = r + 8 * hh;
    float v = acc[r] / sdenom[row];
    v = v > 0.f ? v : OUTSL_ * v;
    out[((long)b * N_ + rb + row) * FOUT_ + ob + nn] = v;
  }
}

// ---------------------------------------------------------------------------
extern "C" void kernel_launch(void* const* d_in, const int* in_sizes, int n_in,
                              void* d_out, int out_size, void* d_ws, size_t ws_size,
                              hipStream_t stream) {
  const float* inp = (const float*)d_in[0];   // (B,N,128) f32
  const int*   adj = (const int*)d_in[1];     // (B,N,N) i32
  const float* W   = (const float*)d_in[2];   // (128,64) f32
  const float* a   = (const float*)d_in[3];   // (128,1)  f32
  float*       out = (float*)d_out;           // (B,N,64) f32

  char* ws = (char*)d_ws;
  __bf16* hb   = (__bf16*)ws;                                     // 2 MB
  float*  srcv = (float*)(ws + (size_t)B_ * N_ * FOUT_ * 2);      // 64 KB
  float*  dstv = srcv + B_ * N_;                                  // 64 KB

  gat_prep<<<B_ * N_, 64, 0, stream>>>(inp, W, a, hb, srcv, dstv);
  gat_attn<<<dim3(N_ / 16, B_), 128, 0, stream>>>(adj, hb, srcv, dstv, out);
}
